// SALSA3D_65163243815507
// MI455X (gfx1250) — matrix operand order repaired and checked by reading, not compiled, not run on hardware
//
#include <hip/hip_runtime.h>
#include <hip/hip_bf16.h>

// ---------------------------------------------------------------------------
// SALSA3D on MI455X (gfx1250, wave32).
// Flash attention (6400x6400, d=144 padded to 160) with v_wmma_f32_16x16x32_f16,
// SPLIT-K over keys (8 segments -> 3200 waves/batch) to fill the SIMDs, then a
// softmax-merge kernel. Q/K/V (~6MB f16) and partials (~33MB) are L2-resident.
// HBM floor is the 2x105MB residual read/write in the fold kernel.
// ---------------------------------------------------------------------------

typedef __attribute__((ext_vector_type(16))) _Float16 v16h;
typedef __attribute__((ext_vector_type(8)))  _Float16 v8h;
typedef __attribute__((ext_vector_type(8)))  float    v8f;

#define HW    160
#define NPIX  (HW * HW)          // 25600
#define NWIN  80
#define NKEY  (NWIN * NWIN)      // 6400
#define LQ    160                // feature dim 144 padded to 160 (5 k-blocks of 32)
#define CIN   64
#define CI    16
#define DDEP  8
#define MID   4
#define SMSCALE 10.0f
#define NSEG  8                  // split-K segments (800 keys each, 25 chunks of 32)
#define SEGLEN (NKEY / NSEG)

static __device__ __forceinline__ v16h cat16(v8h lo, v8h hi) {
  return __builtin_shufflevector(lo, hi, 0,1,2,3,4,5,6,7,8,9,10,11,12,13,14,15);
}

// ---------------------------------------------------------------------------
// Kernel 1: mid-slice 1x1 conv projections (64 -> 16 each for q/v/k sources)
// proj layout: [3][16][160][160] f32  (p=0:g(query) 1:theta(value) 2:phi(key))
// ---------------------------------------------------------------------------
__global__ __launch_bounds__(256) void proj_kernel(
    const float* __restrict__ b,
    const float* __restrict__ gw, const float* __restrict__ gb,
    const float* __restrict__ tw, const float* __restrict__ tb,
    const float* __restrict__ pw, const float* __restrict__ pb,
    float* __restrict__ proj, int batch)
{
  int pix = blockIdx.x * blockDim.x + threadIdx.x;
  if (pix >= NPIX) return;
  float aq[CI], av[CI], ak[CI];
  #pragma unroll
  for (int o = 0; o < CI; ++o) { aq[o] = gb[o]; av[o] = tb[o]; ak[o] = pb[o]; }
  for (int c = 0; c < CIN; ++c) {
    float x = b[(((size_t)batch * CIN + c) * DDEP + MID) * NPIX + pix];
    #pragma unroll
    for (int o = 0; o < CI; ++o) {
      aq[o] = fmaf(gw[o * CIN + c], x, aq[o]);
      av[o] = fmaf(tw[o * CIN + c], x, av[o]);
      ak[o] = fmaf(pw[o * CIN + c], x, ak[o]);
    }
  }
  #pragma unroll
  for (int o = 0; o < CI; ++o) {
    proj[(0 * CI + o) * NPIX + pix] = aq[o];
    proj[(1 * CI + o) * NPIX + pix] = av[o];
    proj[(2 * CI + o) * NPIX + pix] = ak[o];
  }
}

// ---------------------------------------------------------------------------
// Kernel 2: im2col pack to f16. Q,K row-major [6400][160]; V transposed
// [160][6400] so the P*V B-operand reads contiguously. Features 144..159 = 0.
// feature order matches torch unfold reshape: f = ci*9 + kr*3 + kc.
// ---------------------------------------------------------------------------
__global__ __launch_bounds__(256) void pack_kernel(
    const float* __restrict__ proj,
    _Float16* __restrict__ Qh, _Float16* __restrict__ Kh, _Float16* __restrict__ Vt)
{
  int l = blockIdx.x * blockDim.x + threadIdx.x;
  if (l >= NKEY) return;
  int i = l / NWIN, j = l % NWIN;
  for (int f = 0; f < LQ; ++f) {
    float q = 0.f, v = 0.f, k = 0.f;
    if (f < CI * 9) {
      int ci = f / 9, rem = f % 9, kr = rem / 3, kc = rem % 3;
      int h = 2 * i - 1 + kr, w = 2 * j - 1 + kc;
      if (h >= 0 && h < HW && w >= 0 && w < HW) {
        int sp = h * HW + w;
        q = proj[(0 * CI + ci) * NPIX + sp];
        v = proj[(1 * CI + ci) * NPIX + sp];
        k = proj[(2 * CI + ci) * NPIX + sp];
      }
    }
    Qh[(size_t)l * LQ + f] = (_Float16)q;
    Kh[(size_t)l * LQ + f] = (_Float16)k;
    Vt[(size_t)f * NKEY + l] = (_Float16)v;
  }
}

// ---------------------------------------------------------------------------
// Kernel 3: split-K flash attention. 8 waves/block; wave <-> (16 query rows,
// one 800-key segment). 3200 waves per batch.
// A fragment (16x32 f16, ISA 7.12.2): lane = m + 16*khalf; per lane two b128:
//   k in [khalf*8, +8) and [16+khalf*8, +8).
// B fragment (32x16 f16): lane = n + 16*kpart; per lane two b128: k in [kpart*8, +8) and [16+kpart*8, +8).
// C/D (16x16 f32): vgpr r, lane -> (M = r + 8*(lane/16), N = lane%16).
// Emits per-segment unnormalized partials (m, l, O).
// ---------------------------------------------------------------------------
__global__ __launch_bounds__(256) void flash_kernel(
    const _Float16* __restrict__ Q, const _Float16* __restrict__ Kq,
    const _Float16* __restrict__ Vt,
    float* __restrict__ Opart, float* __restrict__ mpart, float* __restrict__ lpart)
{
  __shared__ __align__(16) _Float16 ptile[8][16 * 32];
  const int lane = threadIdx.x & 31;
  const int wave = threadIdx.x >> 5;
  const int widx = blockIdx.x * 8 + wave;      // 0 .. 400*NSEG-1
  const int tile = widx % (NKEY / 16);
  const int seg  = widx / (NKEY / 16);
  const int qbase = tile * 16;
  const int kstart = seg * SEGLEN;
  const int mrow = lane & 15;
  const int hi   = lane >> 4;

  // Q resident as 5 A-fragments (40 VGPRs)
  v16h qf[5];
  {
    const _Float16* qrow = Q + (size_t)(qbase + mrow) * LQ;
    #pragma unroll
    for (int kb = 0; kb < 5; ++kb) {
      const _Float16* p = qrow + kb * 32 + hi * 8;
      qf[kb] = cat16(*(const v8h*)p, *(const v8h*)(p + 16));
    }
  }

  v8f o_acc[10];
  #pragma unroll
  for (int ft = 0; ft < 10; ++ft) o_acc[ft] = (v8f){};
  float m_run[8], l_run[8];
  #pragma unroll
  for (int r = 0; r < 8; ++r) { m_run[r] = -1e30f; l_run[r] = 0.f; }

  _Float16* myP = &ptile[wave][0];

  for (int kc = kstart; kc < kstart + SEGLEN; kc += 32) {
    // ---- S = Q K^T for 32 keys (two 16-key subtiles) ----
    v8f s0 = (v8f){}, s1 = (v8f){};
    const _Float16* krow0 = Kq + (size_t)(kc + mrow) * LQ;
    const _Float16* krow1 = Kq + (size_t)(kc + 16 + mrow) * LQ;
    #pragma unroll
    for (int kb = 0; kb < 5; ++kb) {
      const _Float16* p0 = krow0 + kb * 32 + hi * 8;
      v16h b0 = cat16(*(const v8h*)p0, *(const v8h*)(p0 + 16));
      s0 = __builtin_amdgcn_wmma_f32_16x16x32_f16(false, qf[kb], false, b0,
                                                  (short)0, s0, false, false);
      const _Float16* p1 = krow1 + kb * 32 + hi * 8;
      v16h b1 = cat16(*(const v8h*)p1, *(const v8h*)(p1 + 16));
      s1 = __builtin_amdgcn_wmma_f32_16x16x32_f16(false, qf[kb], false, b1,
                                                  (short)0, s1, false, false);
    }

    // ---- online softmax (row reductions stay inside 16-lane halves) ----
    float p0v[8], p1v[8], sc[8];
    #pragma unroll
    for (int r = 0; r < 8; ++r) {
      float a = s0[r] * SMSCALE, c = s1[r] * SMSCALE;
      float mx = fmaxf(a, c);
      #pragma unroll
      for (int off = 1; off < 16; off <<= 1) mx = fmaxf(mx, __shfl_xor(mx, off, 32));
      float mnew = fmaxf(m_run[r], mx);
      sc[r] = __expf(m_run[r] - mnew);
      m_run[r] = mnew;
      p0v[r] = __expf(a - mnew);
      p1v[r] = __expf(c - mnew);
      float rs = p0v[r] + p1v[r];
      #pragma unroll
      for (int off = 1; off < 16; off <<= 1) rs += __shfl_xor(rs, off, 32);
      l_run[r] = l_run[r] * sc[r] + rs;
    }
    #pragma unroll
    for (int ft = 0; ft < 10; ++ft)
      #pragma unroll
      for (int r = 0; r < 8; ++r) o_acc[ft][r] *= sc[r];

    // ---- P: C/D layout -> A layout through per-wave LDS tile (16x32) ----
    #pragma unroll
    for (int r = 0; r < 8; ++r) {
      int M = r + 8 * hi;
      myP[M * 32 + mrow]      = (_Float16)p0v[r];
      myP[M * 32 + 16 + mrow] = (_Float16)p1v[r];
    }
    asm volatile("s_wait_dscnt 0x0" ::: "memory");  // wave-local LDS RAW
    v16h pf;
    {
      const _Float16* pr = myP + mrow * 32 + hi * 8;
      pf = cat16(*(const v8h*)pr, *(const v8h*)(pr + 16));
    }

    // ---- O += P * V  (V^T rows are contiguous in k) ----
    #pragma unroll
    for (int ft = 0; ft < 10; ++ft) {
      const _Float16* vr = Vt + (size_t)(ft * 16 + mrow) * NKEY + kc + hi * 8;
      v16h vb = cat16(*(const v8h*)vr, *(const v8h*)(vr + 16));
      o_acc[ft] = __builtin_amdgcn_wmma_f32_16x16x32_f16(false, pf, false, vb,
                                                         (short)0, o_acc[ft], false, false);
    }
  }

  // ---- emit unnormalized partials for this segment ----
  #pragma unroll
  for (int ft = 0; ft < 10; ++ft)
    #pragma unroll
    for (int r = 0; r < 8; ++r) {
      int M = r + 8 * hi;
      Opart[((size_t)seg * NKEY + qbase + M) * LQ + ft * 16 + mrow] = o_acc[ft][r];
    }
  if (mrow == 0) {
    #pragma unroll
    for (int r = 0; r < 8; ++r) {
      int M = r + 8 * hi;
      mpart[seg * NKEY + qbase + M] = m_run[r];
      lpart[seg * NKEY + qbase + M] = l_run[r];
    }
  }
}

// ---------------------------------------------------------------------------
// Kernel 3b: merge the NSEG softmax partials per row:
//   M = max_s m_s;  L = sum_s l_s e^{m_s-M};  O = sum_s e^{m_s-M} O_s / L
// ---------------------------------------------------------------------------
__global__ __launch_bounds__(256) void merge_kernel(
    const float* __restrict__ Opart, const float* __restrict__ mpart,
    const float* __restrict__ lpart, float* __restrict__ O)
{
  int idx = blockIdx.x * blockDim.x + threadIdx.x;   // over NKEY*LQ
  if (idx >= NKEY * LQ) return;
  int row = idx / LQ, f = idx % LQ;
  float M = -1e30f;
  #pragma unroll
  for (int s = 0; s < NSEG; ++s) M = fmaxf(M, mpart[s * NKEY + row]);
  float L = 0.f, acc = 0.f;
  #pragma unroll
  for (int s = 0; s < NSEG; ++s) {
    float e = __expf(mpart[s * NKEY + row] - M);
    L += lpart[s * NKEY + row] * e;
    acc += e * Opart[((size_t)s * NKEY + row) * LQ + f];
  }
  O[(size_t)row * LQ + f] = acc / L;
}

// ---------------------------------------------------------------------------
// Kernel 4: fold (overlap-add, analytic overlap counts) + 16->64 conv +
// broadcast over D + residual. This is the HBM-bound 2x105MB tail.
// 1D contribution of coord p: even -> {(p/2, k=1)}; odd<159 -> {((p+1)/2,0),
// ((p-1)/2,2)}; p==159 -> {(79, 2)}.
// ---------------------------------------------------------------------------
__global__ __launch_bounds__(256) void fold_out_kernel(
    const float* __restrict__ O, const float* __restrict__ b,
    const float* __restrict__ Ww, const float* __restrict__ Wb,
    float* __restrict__ out, int batch)
{
  int pix = blockIdx.x * blockDim.x + threadIdx.x;
  if (pix >= NPIX) return;
  int h = pix / HW, w = pix % HW;

  int ih[2], kh[2], nh = 0;
  if (h & 1) {
    if ((h + 1) / 2 < NWIN) { ih[nh] = (h + 1) / 2; kh[nh] = 0; nh++; }
    ih[nh] = (h - 1) / 2; kh[nh] = 2; nh++;
  } else { ih[0] = h / 2; kh[0] = 1; nh = 1; }
  int iw[2], kw[2], nw = 0;
  if (w & 1) {
    if ((w + 1) / 2 < NWIN) { iw[nw] = (w + 1) / 2; kw[nw] = 0; nw++; }
    iw[nw] = (w - 1) / 2; kw[nw] = 2; nw++;
  } else { iw[0] = w / 2; kw[0] = 1; nw = 1; }

  float z[CI];
  #pragma unroll
  for (int ci = 0; ci < CI; ++ci) z[ci] = 0.f;
  for (int a = 0; a < nh; ++a)
    for (int c = 0; c < nw; ++c) {
      int l = ih[a] * NWIN + iw[c];
      const float* src = O + (size_t)l * LQ + kh[a] * 3 + kw[c];
      #pragma unroll
      for (int ci = 0; ci < CI; ++ci) z[ci] += src[ci * 9];
    }
  float invc = 1.0f / (float)(nh * nw);
  #pragma unroll
  for (int ci = 0; ci < CI; ++ci) z[ci] *= invc;

  for (int c = 0; c < CIN; ++c) {
    float acc = Wb[c];
    #pragma unroll
    for (int ci = 0; ci < CI; ++ci) acc = fmaf(Ww[c * CI + ci], z[ci], acc);
    #pragma unroll
    for (int d = 0; d < DDEP; ++d) {
      size_t idx = (((size_t)batch * CIN + c) * DDEP + d) * NPIX + pix;
      out[idx] = b[idx] + acc;
    }
  }
}

// ---------------------------------------------------------------------------
extern "C" void kernel_launch(void* const* d_in, const int* in_sizes, int n_in,
                              void* d_out, int out_size, void* d_ws, size_t ws_size,
                              hipStream_t stream) {
  const float* b   = (const float*)d_in[0];
  const float* gw  = (const float*)d_in[1];
  const float* gb  = (const float*)d_in[2];
  const float* tw  = (const float*)d_in[3];
  const float* tb  = (const float*)d_in[4];
  const float* pw  = (const float*)d_in[5];
  const float* pb  = (const float*)d_in[6];
  const float* Ww  = (const float*)d_in[7];
  const float* Wb  = (const float*)d_in[8];
  float* out = (float*)d_out;

  char* ws = (char*)d_ws;
  // ~48.3 MB, reused across the two (stream-serialized) batch elements.
  size_t off = 0;
  float*    proj  = (float*)   (ws + off); off += (size_t)3 * CI * NPIX * 4;   //  4,915,200
  _Float16* Qh    = (_Float16*)(ws + off); off += (size_t)NKEY * LQ * 2;       // +2,048,000
  _Float16* Kh    = (_Float16*)(ws + off); off += (size_t)NKEY * LQ * 2;
  _Float16* Vt    = (_Float16*)(ws + off); off += (size_t)NKEY * LQ * 2;
  float*    O     = (float*)   (ws + off); off += (size_t)NKEY * LQ * 4;       // +4,096,000
  float*    Opart = (float*)   (ws + off); off += (size_t)NSEG * NKEY * LQ * 4;// +32,768,000
  float*    mpart = (float*)   (ws + off); off += (size_t)NSEG * NKEY * 4;
  float*    lpart = (float*)   (ws + off); off += (size_t)NSEG * NKEY * 4;

  for (int batch = 0; batch < 2; ++batch) {
    proj_kernel<<<(NPIX + 255) / 256, 256, 0, stream>>>(b, gw, gb, tw, tb, pw, pb,
                                                        proj, batch);
    pack_kernel<<<(NKEY + 255) / 256, 256, 0, stream>>>(proj, Qh, Kh, Vt);
    flash_kernel<<<(NKEY / 16) * NSEG / 8, 256, 0, stream>>>(Qh, Kh, Vt,
                                                             Opart, mpart, lpart);
    merge_kernel<<<(NKEY * LQ + 255) / 256, 256, 0, stream>>>(Opart, mpart, lpart, O);
    fold_out_kernel<<<(NPIX + 255) / 256, 256, 0, stream>>>(O, b, Ww, Wb, out, batch);
  }
}
